// Classifier_21191368639252
// MI455X (gfx1250) — compile-verified, matrix operand order checked by reading, not run on hardware
//
#include <hip/hip_runtime.h>
#include <hip/hip_bf16.h>

// PointCNN (XConv x3 + FPS + MLP head) for MI455X / gfx1250.
// All dense _lin layers go through a WMMA f16-in/f32-acc GEMM kernel
// (v_wmma_f32_16x16x32_f16, 4 WMMAs per K-step per wave, 32x32 wave tile)
// with vectorized LDS-staged weight slabs (global_load_b128 -> ds_store_b64),
// branchless K-guards, global prefetch, and a fused bias + ELU/ReLU +
// eval-BatchNorm epilogue.

#define BBATCH 16
#define NPTS   2048
#define NCLS   40
#define BN_EPS 1e-5f

typedef __attribute__((ext_vector_type(16))) _Float16 v16h;
typedef __attribute__((ext_vector_type(4)))  _Float16 v4h;
typedef __attribute__((ext_vector_type(8)))  float    v8f;

__device__ __forceinline__ float eluf(float v) { return v > 0.f ? v : (expf(v) - 1.f); }

// ---------------------------------------------------------------------------
// Generic GEMM: C[M,N] = epilogue(A[M,K] * W[K,N] + bias)
// epilogue: act (0=none,1=elu,2=relu) then optional BN (g,b,m,v per column).
// Block = 256 threads = 8 wave32 waves stacked along M.
// Block tile 256(M) x 32(N); each wave owns a 32x32 tile = 4 accumulators,
// issuing 4 v_wmma_f32_16x16x32_f16 per 32-wide K step.
// NOTE: A loads for k >= K are clamped to index 0 of the row, so there is no
// K-tail over-read past a row.
// ---------------------------------------------------------------------------
__global__ __launch_bounds__(256) void wmma_gemm_kernel(
    const float* __restrict__ A, int M, int K, int lda,
    const float* __restrict__ W, int N,
    const float* __restrict__ bias,
    const float* __restrict__ bng, const float* __restrict__ bnb,
    const float* __restrict__ bnm, const float* __restrict__ bnv,
    float* __restrict__ C, int ldc, int act)
{
  __shared__ _Float16 wt[32][36];   // [k][col]; stride 36 halves = 72B (8B aligned rows)
  const int lane = threadIdx.x & 31;
  const int wave = threadIdx.x >> 5;
  const int row0 = blockIdx.x * 256 + wave * 32;
  const int col0 = blockIdx.y * 32;
  const int lm   = lane & 15;
  const int kb   = (lane >> 4) * 8;         // K sub-offset per lane half
  const int r0   = row0 + lm;               // A row, M-subtile 0
  const int r1   = row0 + 16 + lm;          // A row, M-subtile 1
  // Clamp row pointers in-bounds; OOB-M rows feed only never-stored outputs.
  const float* Ar0 = A + (long)((r0 < M) ? r0 : 0) * lda;
  const float* Ar1 = A + (long)((r1 < M) ? r1 : 0) * lda;

  // Staging coordinates: each of the 256 threads owns one 4-column quad.
  const int skk = threadIdx.x >> 3;          // 0..31 (k row in slab)
  const int sc4 = (threadIdx.x & 7) * 4;     // 0,4,...,28 (col quad)
  const bool n4 = ((N & 3) == 0);            // float4 path legality (uniform)

  v8f acc00 = {0.f, 0.f, 0.f, 0.f, 0.f, 0.f, 0.f, 0.f};
  v8f acc01 = acc00, acc10 = acc00, acc11 = acc00;

  for (int k0 = 0; k0 < K; k0 += 32) {
    __syncthreads();
    // Stage W[k0..k0+31][col0..col0+31] into LDS: one float4 per thread.
    {
      const int kidx = k0 + skk, cidx = col0 + sc4;
      v4h q;
      if (n4 && kidx < K && (cidx + 4) <= N) {
        const float4 wv = *(const float4*)(W + (long)kidx * N + cidx);
        q[0] = (_Float16)wv.x; q[1] = (_Float16)wv.y;
        q[2] = (_Float16)wv.z; q[3] = (_Float16)wv.w;
      } else {
#pragma unroll
        for (int u = 0; u < 4; ++u) {
          bool p = (kidx < K) && ((cidx + u) < N);
          float wv = W[(long)(p ? kidx : 0) * N + (p ? (cidx + u) : 0)];
          q[u] = (_Float16)(p ? wv : 0.f);
        }
      }
      *(v4h*)&wt[skk][sc4] = q;
    }
    __syncthreads();

    if (k0 + 32 < K) {   // prefetch next K slab (speculative, gfx1250 global_prefetch)
      __builtin_prefetch(W + (long)(k0 + 32) * N + col0, 0, 1);
      __builtin_prefetch(Ar0 + k0 + 32, 0, 1);
      __builtin_prefetch(Ar1 + k0 + 32, 0, 1);
    }

    // A fragments (16-bit A 16x32 layout: lanes 0-15 K 0-7 & 16-23,
    // lanes 16-31 K 8-15 & 24-31). K-guard select only (NaN safety);
    // B side is zero for k >= K so valid sums are unaffected.
    v16h a0, a1, b0, b1;
#pragma unroll
    for (int e = 0; e < 16; ++e) {
      int kk = (e < 8) ? (kb + e) : (16 + kb + (e - 8));
      int kidx = k0 + kk;
      bool pk = kidx < K;
      int ks = pk ? kidx : 0;
      float av0 = Ar0[ks];
      float av1 = Ar1[ks];
      a0[e] = (_Float16)(pk ? av0 : 0.f);
      a1[e] = (_Float16)(pk ? av1 : 0.f);
      b0[e] = wt[kk][lm];
      b1[e] = wt[kk][16 + lm];
    }
    acc00 = __builtin_amdgcn_wmma_f32_16x16x32_f16(false, a0, false, b0, (short)0, acc00, false, false);
    acc01 = __builtin_amdgcn_wmma_f32_16x16x32_f16(false, a0, false, b1, (short)0, acc01, false, false);
    acc10 = __builtin_amdgcn_wmma_f32_16x16x32_f16(false, a1, false, b0, (short)0, acc10, false, false);
    acc11 = __builtin_amdgcn_wmma_f32_16x16x32_f16(false, a1, false, b1, (short)0, acc11, false, false);
  }

  // Epilogue. C/D layout: vgpr i, lanes 0-15 -> M=i, lanes 16-31 -> M=i+8.
  const int half = (lane >> 4) * 8;
#pragma unroll
  for (int cg = 0; cg < 2; ++cg) {
    const int cc = col0 + cg * 16 + lm;
    if (cc < N) {
      float g = 1.f, b2 = 0.f, m2 = 0.f, inv = 1.f;
      if (bng) { g = bng[cc]; b2 = bnb[cc]; m2 = bnm[cc]; inv = rsqrtf(bnv[cc] + BN_EPS); }
      const float bi = bias ? bias[cc] : 0.f;
#pragma unroll
      for (int rg = 0; rg < 2; ++rg) {
        v8f acc = (cg == 0) ? (rg == 0 ? acc00 : acc10)
                            : (rg == 0 ? acc01 : acc11);
#pragma unroll
        for (int i = 0; i < 8; ++i) {
          int rr = row0 + rg * 16 + half + i;
          if (rr < M) {
            float v = acc[i] + bi;
            if (act == 1) v = eluf(v);
            else if (act == 2) v = fmaxf(v, 0.f);
            if (bng) v = (v - m2) * g * inv + b2;
            C[(long)rr * ldc + cc] = v;
          }
        }
      }
    }
  }
}

// ---------------------------------------------------------------------------
// kNN with self included, ascending d2, stable (lower index wins ties),
// dilated selection idx[..., ::dil]. One thread per query point.
// ---------------------------------------------------------------------------
__global__ void knn_kernel(const float* __restrict__ pos, int n, int Kd, int dil,
                           int Kout, int* __restrict__ nbr)
{
  long gid = (long)blockIdx.x * blockDim.x + threadIdx.x;
  if (gid >= (long)BBATCH * n) return;
  int b = (int)(gid / n), i = (int)(gid - (long)b * n);
  const float* pb = pos + (long)b * n * 3;
  float xi = pb[3 * i], yi = pb[3 * i + 1], zi = pb[3 * i + 2];
  float dbest[32]; int ibest[32];
  for (int q = 0; q < Kd; ++q) { dbest[q] = 3.0e38f; ibest[q] = 0; }
  for (int j = 0; j < n; ++j) {
    float dx = pb[3 * j] - xi, dy = pb[3 * j + 1] - yi, dz = pb[3 * j + 2] - zi;
    float d2 = dx * dx + dy * dy + dz * dz;
    if (d2 < dbest[Kd - 1]) {
      int q = Kd - 1;
      while (q > 0 && dbest[q - 1] > d2) { dbest[q] = dbest[q - 1]; ibest[q] = ibest[q - 1]; --q; }
      dbest[q] = d2; ibest[q] = j;
    }
  }
  int* out = nbr + ((long)b * n + i) * Kout;
  for (int q = 0; q < Kout; ++q) out[q] = ibest[q * dil];
}

// rel[b,i,kk,:] = pos[b,nbr[b,i,kk],:] - pos[b,i,:]
__global__ void rel_kernel(const float* __restrict__ pos, const int* __restrict__ nbr,
                           int n, int K, float* __restrict__ rel)
{
  long t = (long)blockIdx.x * blockDim.x + threadIdx.x;
  long total = (long)BBATCH * n * K;
  if (t >= total) return;
  long r = t / K; int i = (int)(r % n); int b = (int)(r / n);
  int j = nbr[t];
  const float* pb = pos + (long)b * n * 3;
  float* o = rel + t * 3;
  o[0] = pb[3 * j]     - pb[3 * i];
  o[1] = pb[3 * j + 1] - pb[3 * i + 1];
  o[2] = pb[3 * j + 2] - pb[3 * i + 2];
}

// hcat[..., cd + c] = x[b, nbr, c]  (h part already written by l2 GEMM, ldc=cd+cin)
__global__ void hcat_gather_kernel(const float* __restrict__ x, const int* __restrict__ nbr,
                                   int n, int K, int cd, int cin, float* __restrict__ hcat)
{
  long t = (long)blockIdx.x * blockDim.x + threadIdx.x;
  long total = (long)BBATCH * n * K * cin;
  if (t >= total) return;
  int c = (int)(t % cin); long r = t / cin;      // r = (b*n+i)*K + kk
  long bni = r / K;
  int b = (int)(bni / n);
  int j = nbr[r];
  hcat[r * (cd + cin) + cd + c] = x[((long)b * n + j) * cin + c];
}

// Grouped conv: out[pt,ki,j] = BN(act(sum_t tin[pt,ki,t] * w[ki,j,t] + bias[ki,j]))
__global__ void xform_conv_kernel(const float* __restrict__ tin,
    const float* __restrict__ w, const float* __restrict__ bias,
    const float* __restrict__ g, const float* __restrict__ bb,
    const float* __restrict__ m, const float* __restrict__ v,
    int n, int k, int act, float* __restrict__ tout)
{
  long t = (long)blockIdx.x * blockDim.x + threadIdx.x;
  long total = (long)BBATCH * n * k * k;
  if (t >= total) return;
  int j = (int)(t % k); long r = t / k; int ki = (int)(r % k); long pt = r / k;
  const float* trow = tin + pt * k * k + (long)ki * k;
  const float* wr = w + ((long)ki * k + j) * k;
  float s = bias[ki * k + j];
  for (int tt = 0; tt < k; ++tt) s += trow[tt] * wr[tt];
  if (act) s = eluf(s);
  int ch = ki * k + j;
  s = (s - m[ch]) * g[ch] * rsqrtf(v[ch] + BN_EPS) + bb[ch];
  tout[t] = s;
}

// xt[pt,c,j] = sum_k hcat[pt,k,c] * tmat[pt,k,j]
__global__ void xt_kernel(const float* __restrict__ hcat, const float* __restrict__ tmat,
                          int n, int K, int c, float* __restrict__ xt)
{
  long t = (long)blockIdx.x * blockDim.x + threadIdx.x;
  long total = (long)BBATCH * n * c * K;
  if (t >= total) return;
  int j = (int)(t % K); long r = t / K; int ci = (int)(r % c); long pt = r / c;
  const float* hp = hcat + pt * K * c + ci;
  const float* tp = tmat + pt * K * K + j;
  float s = 0.f;
  for (int kk = 0; kk < K; ++kk) s += hp[(long)kk * c] * tp[(long)kk * K];
  xt[t] = s;
}

// y[pt,ci,d] = sum_j xt[pt,ci,j] * dw_w[ci,d,j] + dw_b[ci,d]
__global__ void dw_kernel(const float* __restrict__ xt, const float* __restrict__ w,
                          const float* __restrict__ bias, int n, int K, int c, int dm,
                          float* __restrict__ y)
{
  long t = (long)blockIdx.x * blockDim.x + threadIdx.x;
  long total = (long)BBATCH * n * c * dm;
  if (t >= total) return;
  int d = (int)(t % dm); long r = t / dm; int ci = (int)(r % c); long pt = r / c;
  const float* xp = xt + (pt * c + ci) * K;
  const float* wp = w + ((long)ci * dm + d) * K;
  float s = bias[ci * dm + d];
  for (int j = 0; j < K; ++j) s += xp[j] * wp[j];
  y[t] = s;
}

// Farthest point sampling, deterministic start at point 0, argmax first-index ties.
__global__ __launch_bounds__(256) void fps_kernel(const float* __restrict__ pos,
                                                  int n, int m, int* __restrict__ out)
{
  __shared__ float d[NPTS];
  __shared__ float rv[256];
  __shared__ int   ri[256];
  int b = blockIdx.x;
  const float* pb = pos + (long)b * n * 3;
  float x0 = pb[0], y0 = pb[1], z0 = pb[2];
  for (int i = threadIdx.x; i < n; i += 256) {
    float dx = pb[3 * i] - x0, dy = pb[3 * i + 1] - y0, dz = pb[3 * i + 2] - z0;
    d[i] = dx * dx + dy * dy + dz * dz;
  }
  if (threadIdx.x == 0) out[(long)b * m] = 0;
  __syncthreads();
  for (int s = 1; s < m; ++s) {
    float best = -1.f; int bi = n;
    for (int i = threadIdx.x; i < n; i += 256)
      if (d[i] > best) { best = d[i]; bi = i; }
    rv[threadIdx.x] = best; ri[threadIdx.x] = bi;
    __syncthreads();
    for (int off = 128; off > 0; off >>= 1) {
      if (threadIdx.x < off) {
        float ov = rv[threadIdx.x + off]; int oi = ri[threadIdx.x + off];
        if (ov > rv[threadIdx.x] || (ov == rv[threadIdx.x] && oi < ri[threadIdx.x])) {
          rv[threadIdx.x] = ov; ri[threadIdx.x] = oi;
        }
      }
      __syncthreads();
    }
    int sel = ri[0];
    if (threadIdx.x == 0) out[(long)b * m + s] = sel;
    __syncthreads();
    float xs = pb[3 * sel], ys = pb[3 * sel + 1], zs = pb[3 * sel + 2];
    for (int i = threadIdx.x; i < n; i += 256) {
      float dx = pb[3 * i] - xs, dy = pb[3 * i + 1] - ys, dz = pb[3 * i + 2] - zs;
      float nd = dx * dx + dy * dy + dz * dz;
      if (nd < d[i]) d[i] = nd;
    }
    __syncthreads();
  }
}

// dst[b,s,c] = src[b, idx[b,s], c]
__global__ void select_kernel(const float* __restrict__ src, const int* __restrict__ idx,
                              int n_src, int m, int C, float* __restrict__ dst)
{
  long t = (long)blockIdx.x * blockDim.x + threadIdx.x;
  long total = (long)BBATCH * m * C;
  if (t >= total) return;
  int c = (int)(t % C); long r = t / C; int s = (int)(r % m); int b = (int)(r / m);
  int j = idx[(long)b * m + s];
  dst[t] = src[((long)b * n_src + j) * C + c];
}

__global__ void mean_pool_kernel(const float* __restrict__ x, int n, int C,
                                 float* __restrict__ out)
{
  int t = blockIdx.x * blockDim.x + threadIdx.x;
  if (t >= BBATCH * C) return;
  int c = t % C, b = t / C;
  float s = 0.f;
  for (int i = 0; i < n; ++i) s += x[((long)b * n + i) * C + c];
  out[t] = s / (float)n;
}

__global__ void log_softmax_kernel(const float* __restrict__ in, float* __restrict__ out, int C)
{
  int b = blockIdx.x * blockDim.x + threadIdx.x;
  if (b >= BBATCH) return;
  const float* r = in + (long)b * C;
  float mx = r[0];
  for (int i = 1; i < C; ++i) mx = fmaxf(mx, r[i]);
  float s = 0.f;
  for (int i = 0; i < C; ++i) s += expf(r[i] - mx);
  float l = logf(s);
  for (int i = 0; i < C; ++i) out[(long)b * C + i] = r[i] - mx - l;
}

// ---------------------------------------------------------------------------
// Host side
// ---------------------------------------------------------------------------
static inline unsigned cdiv(long a, long b) { return (unsigned)((a + b - 1) / b); }

static void gemm_launch(hipStream_t st, const float* A, long M, int K, int lda,
                        const float* W, const float* bias,
                        const float* g, const float* b2, const float* m2, const float* v2,
                        float* C, int N, int ldc, int act)
{
  dim3 grid(cdiv(M, 256), cdiv(N, 32));
  wmma_gemm_kernel<<<grid, 256, 0, st>>>(A, (int)M, K, lda, W, N, bias, g, b2, m2, v2, C, ldc, act);
}

// Param flattening assumption: depth-first in Python dict insertion order.
// Per xconv (34 arrays): l1_w,l1_b, bn1{g,b,m,v}, l2_w,l2_b, bn2{..}, l3_w,l3_b,
// bn3{..}, c1_w,c1_b, bn4{..}, c2_w,c2_b, bn5{..}, dw_w,dw_b, fc_w,fc_b.
static void run_xconv(hipStream_t st, const float* pos, const float* xin, float* xout,
                      int n, int k, int dil, int cd, int cin, int dm, int cout,
                      void* const* d_in, int base,
                      float* REL, float* HX, float* HCAT, float* T0, float* T1, float* Y,
                      int* IDX)
{
  auto P = [&](int i) { return (const float*)d_in[base + i]; };
  const int c = cd + cin;
  const long Mn = (long)BBATCH * n;
  const long Mnk = Mn * k;

  knn_kernel<<<cdiv(Mn, 256), 256, 0, st>>>(pos, n, k * dil, dil, k, IDX);
  rel_kernel<<<cdiv(Mnk, 256), 256, 0, st>>>(pos, IDX, n, k, REL);

  // l1: (Mnk,3) x (3,cd), elu + bn1 -> HX
  gemm_launch(st, REL, Mnk, 3, 3, P(0), P(1), P(2), P(3), P(4), P(5), HX, cd, cd, 1);
  // l2: (Mnk,cd) x (cd,cd), elu + bn2 -> HCAT columns [0,cd)
  gemm_launch(st, HX, Mnk, cd, cd, P(6), P(7), P(8), P(9), P(10), P(11), HCAT, cd, c, 1);
  if (cin > 0)
    hcat_gather_kernel<<<cdiv(Mnk * cin, 256), 256, 0, st>>>(xin, IDX, n, k, cd, cin, HCAT);

  // l3: (Mn, 3k) x (3k, k*k), elu + bn3 -> T0
  gemm_launch(st, REL, Mn, 3 * k, 3 * k, P(12), P(13), P(14), P(15), P(16), P(17),
              T0, k * k, k * k, 1);
  xform_conv_kernel<<<cdiv(Mn * k * k, 256), 256, 0, st>>>(
      T0, P(18), P(19), P(20), P(21), P(22), P(23), n, k, 1, T1);
  xform_conv_kernel<<<cdiv(Mn * k * k, 256), 256, 0, st>>>(
      T1, P(24), P(25), P(26), P(27), P(28), P(29), n, k, 0, T0);

  xt_kernel<<<cdiv(Mn * c * k, 256), 256, 0, st>>>(HCAT, T0, n, k, c, HX);
  dw_kernel<<<cdiv(Mn * c * dm, 256), 256, 0, st>>>(HX, P(30), P(31), n, k, c, dm, Y);

  // fc: (Mn, c*dm) x (c*dm, cout), + outer relu -> xout
  gemm_launch(st, Y, Mn, c * dm, c * dm, P(32), P(33),
              nullptr, nullptr, nullptr, nullptr, xout, cout, cout, 2);
}

extern "C" void kernel_launch(void* const* d_in, const int* in_sizes, int n_in,
                              void* d_out, int out_size, void* d_ws, size_t ws_size,
                              hipStream_t stream)
{
  (void)in_sizes; (void)n_in; (void)out_size; (void)ws_size;
  const float* pos = (const float*)d_in[0];   // (B*N, 3)
  // d_in[1] = batch (unused: dense equal-sized clouds)
  // bases: xc1=2, xc2=36, xc3=70, head lin weights 104..109

  // Workspace layout (floats)
  float* ws   = (float*)d_ws;
  float* REL  = ws;                     // <= 16*2048*8*3       = 786432
  float* HX   = REL  + 786432;          // lifted feats / xt    = 16515072
  float* HCAT = HX   + 16515072;        // [h | gathered x]     = 16515072
  float* T0   = HCAT + 16515072;        // transform ping       = 2097152
  float* T1   = T0   + 2097152;         // transform pong       = 2097152
  float* Y    = T1   + 2097152;         // depthwise output     = 2097152
  float* XA   = Y    + 2097152;         // features ping        = 1572864
  float* XB   = XA   + 1572864;         // features pong        = 1572864
  float* POS2 = XB   + 1572864;         // 16*768*3
  float* POS3 = POS2 + 36864;           // 16*256*3
  float* POOL = POS3 + 12288;           // 16*192
  float* H1   = POOL + 3072;            // 16*256
  float* H2   = H1   + 4096;            // 16*128
  float* LOGI = H2   + 2048;            // 16*40
  int*   IDX  = (int*)(LOGI + 640);     // <= 16*2048*8 neighbor ids
  int*   FPSI = IDX + 262144;           // <= 16*768 sample ids

  // ---- Level 1: n=2048, k=8, dil=1, cd=32, cin=0, dm=2, cout=48 ----
  run_xconv(stream, pos, nullptr, XA, 2048, 8, 1, 32, 0, 2, 48,
            d_in, 2, REL, HX, HCAT, T0, T1, Y, IDX);

  // FPS 2048 -> 768, select pos/features
  fps_kernel<<<BBATCH, 256, 0, stream>>>(pos, 2048, 768, FPSI);
  select_kernel<<<cdiv((long)BBATCH * 768 * 3, 256), 256, 0, stream>>>(pos, FPSI, 2048, 768, 3, POS2);
  select_kernel<<<cdiv((long)BBATCH * 768 * 48, 256), 256, 0, stream>>>(XA, FPSI, 2048, 768, 48, XB);

  // ---- Level 2: n=768, k=12, dil=2, cd=64, cin=48, dm=1, cout=96 ----
  run_xconv(stream, POS2, XB, XA, 768, 12, 2, 64, 48, 1, 96,
            d_in, 36, REL, HX, HCAT, T0, T1, Y, IDX);

  // FPS 768 -> 256
  fps_kernel<<<BBATCH, 256, 0, stream>>>(POS2, 768, 256, FPSI);
  select_kernel<<<cdiv((long)BBATCH * 256 * 3, 256), 256, 0, stream>>>(POS2, FPSI, 768, 256, 3, POS3);
  select_kernel<<<cdiv((long)BBATCH * 256 * 96, 256), 256, 0, stream>>>(XA, FPSI, 768, 256, 96, XB);

  // ---- Level 3: n=256, k=16, dil=2, cd=128, cin=96, dm=1, cout=192 ----
  run_xconv(stream, POS3, XB, XA, 256, 16, 2, 128, 96, 1, 192,
            d_in, 70, REL, HX, HCAT, T0, T1, Y, IDX);

  // ---- Head: mean pool -> 192 -> 256 -> 128 -> 40 -> log_softmax ----
  mean_pool_kernel<<<cdiv(BBATCH * 192, 256), 256, 0, stream>>>(XA, 256, 192, POOL);
  gemm_launch(stream, POOL, BBATCH, 192, 192, (const float*)d_in[104], (const float*)d_in[105],
              nullptr, nullptr, nullptr, nullptr, H1, 256, 256, 2);
  gemm_launch(stream, H1, BBATCH, 256, 256, (const float*)d_in[106], (const float*)d_in[107],
              nullptr, nullptr, nullptr, nullptr, H2, 128, 128, 2);
  gemm_launch(stream, H2, BBATCH, 128, 128, (const float*)d_in[108], (const float*)d_in[109],
              nullptr, nullptr, nullptr, nullptr, LOGI, NCLS, NCLS, 0);
  log_softmax_kernel<<<1, 32, 0, stream>>>(LOGI, (float*)d_out, NCLS);
}
